// Local_SS2D_62045097558410
// MI455X (gfx1250) — compile-verified, matrix operand order checked by reading, not run on hardware
//
#include <hip/hip_runtime.h>
#include <hip/hip_bf16.h>
#include <cstdint>

#define BB    8
#define LL    4096
#define DM    64
#define DI    128
#define MALL  (BB*LL)   // 32768

typedef __attribute__((ext_vector_type(16))) _Float16 v16h;
typedef __attribute__((ext_vector_type(8)))  _Float16 v8h;
typedef __attribute__((ext_vector_type(8)))  float    v8f;

// ---------------------------------------------------------------- CDNA5 async copy
// GLOBAL_LOAD_ASYNC_TO_LDS_B128: VDST = VGPR holding LDS byte address,
// VADDR = 64-bit global address (GV mode). Tracked by ASYNCcnt.
__device__ __forceinline__ void async_copy16(unsigned lds_addr, const _Float16* g) {
  uint64_t ga = (uint64_t)(uintptr_t)g;
  asm volatile("global_load_async_to_lds_b128 %0, %1, off"
               :: "v"(lds_addr), "v"(ga) : "memory");
}
__device__ __forceinline__ void wait_async() {
  asm volatile("s_wait_asynccnt 0x0" ::: "memory");
}

// ---------------------------------------------------------------- WMMA helpers
// A (16x32 f16) and B (32x16 f16) fragments share the same per-lane layout:
// row = lane&15 ; element j holds K = (j>>3)*16 + (lane>>4)*8 + (j&7)
// => two contiguous 16-byte chunks at k0 + (lane>>4)*8 and k0 + 16 + (lane>>4)*8.
__device__ __forceinline__ v16h ldfrag(const _Float16* __restrict__ base, int ld,
                                       int row, int k0, int hh) {
  const _Float16* p = base + (size_t)row * ld + k0 + hh * 8;
  v8h lo = *(const v8h*)p;
  v8h hi = *(const v8h*)(p + 16);
  return __builtin_shufflevector(lo, hi, 0,1,2,3,4,5,6,7,8,9,10,11,12,13,14,15);
}

__device__ __forceinline__ v8f wmma16(v16h a, v16h b, v8f c) {
  return __builtin_amdgcn_wmma_f32_16x16x32_f16(false, a, false, b, (short)0, c,
                                                false, false);
}

// scan-order -> row-major position map per direction
__device__ __forceinline__ int qmap(int k, int l) {
  switch (k & 3) {
    case 0:  return l;
    case 1:  return ((l & 63) << 6) | (l >> 6);
    case 2:  return 4095 - l;
    default: { int m = 4095 - l; return ((m & 63) << 6) | (m >> 6); }
  }
}

// ---------------------------------------------------------------- weight convert
// Wip [256,64] f16 ; Wx [2 pairs][2 dirs][48 rows][128] f16 (rows 36..47 zero);
// Wop [64,128] f16
__global__ void k_convert_w(const float* __restrict__ in_proj_w,
                            const float* __restrict__ x_proj_w,
                            const float* __restrict__ out_proj_w,
                            _Float16* __restrict__ Wip,
                            _Float16* __restrict__ Wx,
                            _Float16* __restrict__ Wop) {
  int i = blockIdx.x * 256 + threadIdx.x;
  if (i < 16384) {
    Wip[i] = (_Float16)in_proj_w[i];
  } else if (i < 16384 + 24576) {
    int j = i - 16384;
    int d = j & 127;
    int row = j >> 7;                 // row = p*96 + jj*48 + c
    int p = row / 96, rem = row % 96;
    int jj = rem / 48, c = rem % 48;
    int dir = p + 2 * jj;
    Wx[j] = (c < 36) ? (_Float16)x_proj_w[(dir * 36 + c) * 128 + d] : (_Float16)0.f;
  } else {
    int j = i - 40960;
    Wop[j] = (_Float16)out_proj_w[j];
  }
}

// ---------------------------------------------------------------- pre layernorm
// x [B,128,L] -> x1n [B,64,L] f32 ; X2h [B*L, 64] f16 (pixel-major)
__global__ void k_preln(const float* __restrict__ x,
                        const float* __restrict__ pre_w,
                        const float* __restrict__ pre_b,
                        float* __restrict__ x1n, _Float16* __restrict__ X2h) {
  int p = blockIdx.x * 256 + threadIdx.x;
  int b = p >> 12, pl = p & 4095;
  const float* xb = x + (size_t)b * 128 * LL + pl;
  float s = 0.f, ss = 0.f;
  for (int c = 0; c < 128; ++c) { float v = xb[(size_t)c * LL]; s += v; ss += v * v; }
  float mu = s * (1.f / 128.f);
  float var = ss * (1.f / 128.f) - mu * mu;
  float rinv = rsqrtf(var + 1e-6f);
  for (int c = 0; c < 128; ++c) {
    float v = (xb[(size_t)c * LL] - mu) * rinv * pre_w[c] + pre_b[c];
    if (c < 64) x1n[(b * 64 + c) * LL + pl] = v;
    else        X2h[(size_t)p * 64 + (c - 64)] = (_Float16)v;
  }
}

// ---------------------------------------------------------------- depthwise 3x3
// in [B,C,L]; optional bias/silu; optional f16 pixel-major & transposed copies.
__global__ void k_dwconv(const float* __restrict__ in, const float* __restrict__ w,
                         const float* __restrict__ bias, float* __restrict__ out,
                         _Float16* __restrict__ pm, _Float16* __restrict__ tm,
                         int C, int do_silu) {
  int idx = blockIdx.x * 256 + threadIdx.x;    // over B*C*L
  int pl = idx & 4095;
  int bc = idx >> 12;
  int c = bc % C, b = bc / C;
  int h = pl >> 6, wc = pl & 63;
  const float* wp = w + c * 9;
  const float* inp = in + (size_t)bc * LL;
  float acc = 0.f;
#pragma unroll
  for (int dy = -1; dy <= 1; ++dy)
#pragma unroll
    for (int dx = -1; dx <= 1; ++dx) {
      int hh = h + dy, ww = wc + dx;
      if (hh >= 0 && hh < 64 && ww >= 0 && ww < 64)
        acc += wp[(dy + 1) * 3 + (dx + 1)] * inp[hh * 64 + ww];
    }
  if (bias) acc += bias[c];
  if (do_silu) acc = acc / (1.f + __expf(-acc));
  out[(size_t)bc * LL + pl] = acc;
  if (pm) pm[((size_t)b * LL + pl) * 128 + c] = (_Float16)acc;
  if (tm) tm[((size_t)b * LL + (wc * 64 + h)) * 128 + c] = (_Float16)acc;
}

// ---------------------------------------------------------------- in_proj GEMM
// X2h[32768,64] @ Wip[256,64]^T ; cols<128 -> xx [B,128,L], cols>=128 -> z [B*L,128]
// 256 thr = 8 waves; weights (64 rows x 64 K = 8KB) async-staged into LDS once.
__global__ void __launch_bounds__(256) k_gemm_inproj(const _Float16* __restrict__ A,
                                                     const _Float16* __restrict__ W,
                                                     float* __restrict__ xx,
                                                     float* __restrict__ z) {
  __shared__ _Float16 sW[64 * 64];
  int tid = threadIdx.x;
  int n0g = blockIdx.y * 64;
  {
    unsigned lbase = (unsigned)(uintptr_t)sW;
    const _Float16* Wblk = W + (size_t)n0g * 64;   // contiguous 64x64 block
#pragma unroll
    for (int i = 0; i < 2; ++i) {
      int idx = tid + i * 256;
      async_copy16(lbase + idx * 16, Wblk + idx * 8);
    }
    wait_async();
  }
  __syncthreads();

  int wave = tid >> 5, lane = tid & 31;
  int hh = lane >> 4, ln = lane & 15;
  int m0 = (blockIdx.x * 8 + wave) * 16;
  v8f acc[4] = {};
#pragma unroll
  for (int ks = 0; ks < 2; ++ks) {
    v16h a = ldfrag(A, 64, m0 + ln, ks * 32, hh);
#pragma unroll
    for (int t = 0; t < 4; ++t) {
      v16h bf = ldfrag(sW, 64, t * 16 + ln, ks * 32, hh);
      acc[t] = wmma16(a, bf, acc[t]);
    }
  }
#pragma unroll
  for (int t = 0; t < 4; ++t)
#pragma unroll
    for (int r = 0; r < 8; ++r) {
      int m = m0 + r + 8 * hh;
      int col = n0g + t * 16 + ln;
      float v = acc[t][r];
      int b = m >> 12, p = m & 4095;
      if (col < 128) xx[((size_t)b * 128 + col) * LL + p] = v;
      else           z[(size_t)m * 128 + (col - 128)] = v;
    }
}

// ---------------------------------------------------------------- x_dbl GEMM
// pair 0: A=u_pm (dirs 0,2) ; pair 1: A=u_tm (dirs 1,3). W block [96,128] (24KB)
// async-staged to LDS; 8 waves x (16M x 96N) tiles per block.
// out: xdbl[b][dir][q][48]  (q = row-major position the row corresponds to)
__global__ void __launch_bounds__(256) k_gemm_xdbl(const _Float16* __restrict__ upm,
                                                   const _Float16* __restrict__ utm,
                                                   const _Float16* __restrict__ Wx,
                                                   float* __restrict__ xdbl) {
  __shared__ _Float16 sW[96 * 128];
  int tid = threadIdx.x;
  int bz = blockIdx.z, b = bz >> 1, pair = bz & 1;
  {
    unsigned lbase = (unsigned)(uintptr_t)sW;
    const _Float16* Wp = Wx + (size_t)pair * 96 * 128;  // contiguous 96x128
#pragma unroll
    for (int i = 0; i < 6; ++i) {
      int idx = tid + i * 256;
      async_copy16(lbase + idx * 16, Wp + idx * 8);
    }
    wait_async();
  }
  __syncthreads();

  int wave = tid >> 5, lane = tid & 31;
  int hh = lane >> 4, ln = lane & 15;
  const _Float16* Ab = (pair ? utm : upm) + (size_t)b * LL * 128;
  int m0 = (blockIdx.x * 8 + wave) * 16;
  v8f acc[6] = {};
#pragma unroll
  for (int ks = 0; ks < 4; ++ks) {
    v16h a = ldfrag(Ab, 128, m0 + ln, ks * 32, hh);
    if (ks < 3) __builtin_prefetch(Ab + (size_t)(m0 + ln) * 128 + (ks + 1) * 32, 0, 1);
#pragma unroll
    for (int t = 0; t < 6; ++t) {
      v16h bf = ldfrag(sW, 128, t * 16 + ln, ks * 32, hh);
      acc[t] = wmma16(a, bf, acc[t]);
    }
  }
#pragma unroll
  for (int t = 0; t < 6; ++t)
#pragma unroll
    for (int r = 0; r < 8; ++r) {
      int m = m0 + r + 8 * hh;
      int q = pair ? (((m & 63) << 6) | (m >> 6)) : m;
      int n = t * 16 + ln;
      int dir = pair + 2 * (n / 48);
      int cc = n % 48;
      xdbl[(((size_t)(b * 4 + dir) * LL + q) * 48) + cc] = acc[t][r];
    }
}

// ---------------------------------------------------------------- selective scan
// grid = B*K (32 blocks) x 256 thr (8 waves x 16 ch, 2 lanes/ch x 8 states)
__global__ void k_scan(const float* __restrict__ xdbl, const float* __restrict__ u,
                       const float* __restrict__ dt_w, const float* __restrict__ dt_b,
                       const float* __restrict__ A_logs, const float* __restrict__ Ds,
                       float* __restrict__ y4) {
  int bk = blockIdx.x, b = bk >> 2, k = bk & 3;
  int tid = threadIdx.x, wave = tid >> 5, lane = tid & 31;
  int c = wave * 16 + (lane & 15);
  int nh = lane >> 4, n0 = nh * 8;
  int kc = k * 128 + c;
  float Av[8], dw4[4];
#pragma unroll
  for (int j = 0; j < 8; ++j) Av[j] = -__expf(A_logs[kc * 16 + n0 + j]);
#pragma unroll
  for (int r = 0; r < 4; ++r) dw4[r] = dt_w[kc * 4 + r];
  float dtb = dt_b[kc], Dk = Ds[kc];
  float h[8] = {0, 0, 0, 0, 0, 0, 0, 0};
  const float* xb = xdbl + (size_t)bk * LL * 48;
  const float* ub = u + ((size_t)b * 128 + c) * LL;
#pragma unroll 1
  for (int l = 0; l < LL; ++l) {
    int q = qmap(k, l);
    const float* xr = xb + (size_t)q * 48;
    if (l + 8 < LL) {
      int qp = qmap(k, l + 8);
      __builtin_prefetch(xb + (size_t)qp * 48, 0, 1);
      __builtin_prefetch(ub + qp, 0, 1);
    }
    float draw = dtb + dw4[0] * xr[0] + dw4[1] * xr[1] + dw4[2] * xr[2] + dw4[3] * xr[3];
    float delta = (draw > 20.f) ? draw : log1pf(__expf(draw));
    float uval = ub[q];
    float du = delta * uval;
    float yp = 0.f;
#pragma unroll
    for (int j = 0; j < 8; ++j) {
      float Bv = xr[4 + n0 + j];
      float Cv = xr[20 + n0 + j];
      h[j] = h[j] * __expf(delta * Av[j]) + du * Bv;
      yp += h[j] * Cv;
    }
    yp += __shfl_xor(yp, 16, 32);
    if (nh == 0)
      y4[(((size_t)b * LL + q) * 128 + c) * 4 + k] = yp + Dk * uval;
  }
}

// ---------------------------------------------------------------- out norm+gate
// wave per pixel: sum 4 direction planes, layernorm(128), * silu(z) -> f16
__global__ void k_outnorm(const float* __restrict__ y4, const float* __restrict__ z,
                          const float* __restrict__ wgt, const float* __restrict__ bia,
                          _Float16* __restrict__ Yh) {
  int wid = threadIdx.x >> 5, lane = threadIdx.x & 31;
  int p = blockIdx.x * 8 + wid;
  float vals[4], s = 0.f, ss = 0.f;
#pragma unroll
  for (int j = 0; j < 4; ++j) {
    int c = lane * 4 + j;
    float4 f = *(const float4*)(y4 + ((size_t)p * 128 + c) * 4);
    float v = f.x + f.y + f.z + f.w;
    vals[j] = v; s += v; ss += v * v;
  }
#pragma unroll
  for (int off = 16; off >= 1; off >>= 1) {
    s += __shfl_xor(s, off, 32);
    ss += __shfl_xor(ss, off, 32);
  }
  float mu = s * (1.f / 128.f);
  float var = ss * (1.f / 128.f) - mu * mu;
  float rinv = rsqrtf(var + 1e-5f);
#pragma unroll
  for (int j = 0; j < 4; ++j) {
    int c = lane * 4 + j;
    float g = (vals[j] - mu) * rinv * wgt[c] + bia[c];
    float zv = z[(size_t)p * 128 + c];
    float sl = zv / (1.f + __expf(-zv));
    Yh[(size_t)p * 128 + c] = (_Float16)(g * sl);
  }
}

// ---------------------------------------------------------------- out_proj GEMM
// Yh[32768,128] @ Wop[64,128]^T -> x2res[32768,64]; Wop (16KB) async-staged to LDS.
__global__ void __launch_bounds__(256) k_gemm_outproj(const _Float16* __restrict__ A,
                                                      const _Float16* __restrict__ W,
                                                      float* __restrict__ x2res) {
  __shared__ _Float16 sW[64 * 128];
  int tid = threadIdx.x;
  {
    unsigned lbase = (unsigned)(uintptr_t)sW;
#pragma unroll
    for (int i = 0; i < 4; ++i) {
      int idx = tid + i * 256;
      async_copy16(lbase + idx * 16, W + idx * 8);
    }
    wait_async();
  }
  __syncthreads();

  int wave = tid >> 5, lane = tid & 31;
  int hh = lane >> 4, ln = lane & 15;
  int m0 = (blockIdx.x * 8 + wave) * 16;
  v8f acc[4] = {};
#pragma unroll
  for (int ks = 0; ks < 4; ++ks) {
    v16h a = ldfrag(A, 128, m0 + ln, ks * 32, hh);
    if (ks < 3) __builtin_prefetch(A + (size_t)(m0 + ln) * 128 + (ks + 1) * 32, 0, 1);
#pragma unroll
    for (int t = 0; t < 4; ++t) {
      v16h bf = ldfrag(sW, 128, t * 16 + ln, ks * 32, hh);
      acc[t] = wmma16(a, bf, acc[t]);
    }
  }
#pragma unroll
  for (int t = 0; t < 4; ++t)
#pragma unroll
    for (int r = 0; r < 8; ++r) {
      int m = m0 + r + 8 * hh;
      x2res[(size_t)m * 64 + t * 16 + ln] = acc[t][r];
    }
}

// ---------------------------------------------------------------- final interleave + LN
__global__ void k_final(const float* __restrict__ x1c, const float* __restrict__ x2res,
                        const float* __restrict__ post_w, const float* __restrict__ post_b,
                        float* __restrict__ out) {
  int p = blockIdx.x * 256 + threadIdx.x;
  int b = p >> 12, pl = p & 4095;
  float s = 0.f, ss = 0.f;
  for (int c2 = 0; c2 < 128; ++c2) {
    float v = (c2 & 1) ? x2res[(size_t)p * 64 + (c2 >> 1)]
                       : x1c[((size_t)b * 64 + (c2 >> 1)) * LL + pl];
    s += v; ss += v * v;
  }
  float mu = s * (1.f / 128.f);
  float var = ss * (1.f / 128.f) - mu * mu;
  float rinv = rsqrtf(var + 1e-6f);
  for (int c2 = 0; c2 < 128; ++c2) {
    float v = (c2 & 1) ? x2res[(size_t)p * 64 + (c2 >> 1)]
                       : x1c[((size_t)b * 64 + (c2 >> 1)) * LL + pl];
    out[((size_t)b * 128 + c2) * LL + pl] = (v - mu) * rinv * post_w[c2] + post_b[c2];
  }
}

// ================================================================ launch
extern "C" void kernel_launch(void* const* d_in, const int* in_sizes, int n_in,
                              void* d_out, int out_size, void* d_ws, size_t ws_size,
                              hipStream_t stream) {
  const float* x          = (const float*)d_in[0];
  const float* pre_w      = (const float*)d_in[1];
  const float* pre_b      = (const float*)d_in[2];
  const float* post_w     = (const float*)d_in[3];
  const float* post_b     = (const float*)d_in[4];
  const float* dw_w       = (const float*)d_in[5];
  const float* in_proj_w  = (const float*)d_in[6];
  const float* conv_w     = (const float*)d_in[7];
  const float* conv_b     = (const float*)d_in[8];
  const float* x_proj_w   = (const float*)d_in[9];
  const float* dt_w       = (const float*)d_in[10];
  const float* dt_b       = (const float*)d_in[11];
  const float* A_logs     = (const float*)d_in[12];
  const float* Ds         = (const float*)d_in[13];
  const float* out_norm_w = (const float*)d_in[14];
  const float* out_norm_b = (const float*)d_in[15];
  const float* out_proj_w = (const float*)d_in[16];
  float* out = (float*)d_out;

  char* wsb = (char*)d_ws;
  size_t off = 0;
  auto alloc = [&](size_t bytes) -> void* {
    void* p = wsb + off;
    off = (off + bytes + 255) & ~(size_t)255;
    return p;
  };
  _Float16* X2h  = (_Float16*)alloc((size_t)MALL * 64 * 2);     // 4 MB
  _Float16* Wip  = (_Float16*)alloc(256 * 64 * 2);
  _Float16* Wx   = (_Float16*)alloc(4 * 48 * 128 * 2);
  _Float16* Wop  = (_Float16*)alloc(64 * 128 * 2);
  float*    xx   = (float*)alloc((size_t)BB * 128 * LL * 4);    // 16 MB
  float*    z    = (float*)alloc((size_t)MALL * 128 * 4);       // 16 MB
  float*    x1n  = (float*)alloc((size_t)BB * 64 * LL * 4);     // 8 MB
  float*    x1c  = (float*)alloc((size_t)BB * 64 * LL * 4);     // 8 MB
  float*    u    = (float*)alloc((size_t)BB * 128 * LL * 4);    // 16 MB
  _Float16* upm  = (_Float16*)alloc((size_t)MALL * 128 * 2);    // 8 MB
  _Float16* utm  = (_Float16*)alloc((size_t)MALL * 128 * 2);    // 8 MB
  float*    xdbl = (float*)alloc((size_t)BB * 4 * LL * 48 * 4); // 25 MB
  float*    y4   = (float*)alloc((size_t)MALL * 128 * 4 * 4);   // 64 MB
  _Float16* Yh   = (_Float16*)alloc((size_t)MALL * 128 * 2);    // 8 MB
  float*    x2r  = (float*)alloc((size_t)MALL * 64 * 4);        // 8 MB

  k_convert_w<<<192, 256, 0, stream>>>(in_proj_w, x_proj_w, out_proj_w, Wip, Wx, Wop);
  k_preln<<<MALL / 256, 256, 0, stream>>>(x, pre_w, pre_b, x1n, X2h);
  k_gemm_inproj<<<dim3(MALL / 128, 4), 256, 0, stream>>>(X2h, Wip, xx, z);
  k_dwconv<<<BB * 64 * LL / 256, 256, 0, stream>>>(x1n, dw_w, nullptr, x1c,
                                                   nullptr, nullptr, 64, 0);
  k_dwconv<<<BB * 128 * LL / 256, 256, 0, stream>>>(xx, conv_w, conv_b, u,
                                                    upm, utm, 128, 1);
  k_gemm_xdbl<<<dim3(LL / 128, 1, BB * 2), 256, 0, stream>>>(upm, utm, Wx, xdbl);
  k_scan<<<BB * 4, 256, 0, stream>>>(xdbl, u, dt_w, dt_b, A_logs, Ds, y4);
  k_outnorm<<<MALL / 8, 256, 0, stream>>>(y4, z, out_norm_w, out_norm_b, Yh);
  k_gemm_outproj<<<MALL / 128, 256, 0, stream>>>(Yh, Wop, x2r);
  k_final<<<MALL / 256, 256, 0, stream>>>(x1c, x2r, post_w, post_b, out);
}
